// SubjectLayers_77214922047957
// MI455X (gfx1250) — matrix operand order repaired and checked by reading, not compiled, not run on hardware
//
#include <hip/hip_runtime.h>

// ---- CDNA5 WMMA vector types ----
typedef __attribute__((ext_vector_type(16))) __bf16 bf16x16;
typedef __attribute__((ext_vector_type(2)))  __bf16 bf16x2;
typedef __attribute__((ext_vector_type(8)))  float  f32x8;

#define B_DIM 256
#define T_DIM 512
#define K_DIM 256
#define N_DIM 256

#define BT 128          // T rows per block
#define BN 128          // OUT cols per block
#define KC 128          // K chunk staged in LDS
#define LDK 136         // padded LDS row stride (bf16 elements): KC + 8

struct Frag32B { uint4 q0, q1; };   // 32 bytes -> bf16x16

// RNE f32 -> bf16 (for the lo term)
static __device__ __forceinline__ unsigned short f2bf(float f) {
  unsigned u = __float_as_uint(f);
  u += 0x7FFFu + ((u >> 16) & 1u);
  return (unsigned short)(u >> 16);
}

// pack two f32 -> two bf16 (RNE); use HW packed convert when available
static __device__ __forceinline__ unsigned pack_bf16_rne(float a, float b) {
#if __has_builtin(__builtin_amdgcn_cvt_pk_bf16_f32)
  bf16x2 r = __builtin_amdgcn_cvt_pk_bf16_f32(a, b);
  return __builtin_bit_cast(unsigned, r);
#else
  return (unsigned)f2bf(a) | ((unsigned)f2bf(b) << 16);
#endif
}

__global__ __launch_bounds__(256)
void subject_layers_gemm(const float* __restrict__ x,
                         const int* __restrict__ sid,
                         const float* __restrict__ w,
                         const float* __restrict__ bias,
                         float* __restrict__ out)
{
  // Split-precision planes: A row-major [row][k], B transposed [n][k]
  __shared__ unsigned short Ahi[BT * LDK];
  __shared__ unsigned short Alo[BT * LDK];
  __shared__ unsigned short Bhi[BN * LDK];
  __shared__ unsigned short Blo[BN * LDK];

  const int tid = threadIdx.x;
  const int bid = blockIdx.x;
  const int b   = bid >> 3;        // batch
  const int rem = bid & 7;
  const int tT  = rem >> 1;        // 0..3  (128 rows of T each)
  const int nT  = rem & 1;         // 0..1  (128 cols) — fastest so x-tile hits L2 twice
  const int s   = sid[b];

  const float* xg = x    + ((size_t)b * T_DIM + (size_t)tT * BT) * K_DIM;
  const float* wg = w    + (size_t)s * (K_DIM * N_DIM) + (size_t)nT * BN;
  const float* bg = bias + (size_t)s * N_DIM + (size_t)nT * BN;
  float*       og = out  + ((size_t)b * T_DIM + (size_t)tT * BT) * N_DIM + (size_t)nT * BN;

  const int lane  = tid & 31;
  const int wave  = tid >> 5;
  const int wm    = wave >> 1;     // 0..3 -> 32 output rows each
  const int wn    = wave & 1;      // 0..1 -> 64 output cols each
  const int m0    = wm * 32;
  const int n0    = wn * 64;
  const int lhalf = lane >> 4;     // lane half selects K sub-range per ISA layout
  const int l15   = lane & 15;

  f32x8 acc[2][4];
  #pragma unroll
  for (int i = 0; i < 2; ++i)
    #pragma unroll
    for (int j = 0; j < 4; ++j)
      #pragma unroll
      for (int r = 0; r < 8; ++r)
        acc[i][j][r] = 0.0f;

  for (int kc = 0; kc < K_DIM; kc += KC) {
    // ---- stage X tile (BT x KC): hi = truncate(f32), lo = RNE(f - hi) ----
    #pragma unroll
    for (int it = 0; it < 16; ++it) {
      int idx = tid + it * 256;            // 0..4095 float4 slots
      int row = idx >> 5;
      int kq  = idx & 31;
      float4 v = *(const float4*)(xg + (size_t)row * K_DIM + kc + kq * 4);
      unsigned u0 = __float_as_uint(v.x), u1 = __float_as_uint(v.y);
      unsigned u2 = __float_as_uint(v.z), u3 = __float_as_uint(v.w);
      uint2 ph = make_uint2((u0 >> 16) | (u1 & 0xFFFF0000u),
                            (u2 >> 16) | (u3 & 0xFFFF0000u));
      float lo0 = v.x - __uint_as_float(u0 & 0xFFFF0000u);
      float lo1 = v.y - __uint_as_float(u1 & 0xFFFF0000u);
      float lo2 = v.z - __uint_as_float(u2 & 0xFFFF0000u);
      float lo3 = v.w - __uint_as_float(u3 & 0xFFFF0000u);
      uint2 pl = make_uint2(pack_bf16_rne(lo0, lo1), pack_bf16_rne(lo2, lo3));
      *(uint2*)&Ahi[row * LDK + kq * 4] = ph;
      *(uint2*)&Alo[row * LDK + kq * 4] = pl;
    }
    // ---- stage W tile (KC x BN), transposed into [n][k] hi/lo planes ----
    #pragma unroll
    for (int it = 0; it < 16; ++it) {
      int idx = tid + it * 256;
      int k   = idx >> 5;
      int nq  = idx & 31;
      float4 v = *(const float4*)(wg + (size_t)(kc + k) * N_DIM + nq * 4);
      float f[4] = {v.x, v.y, v.z, v.w};
      #pragma unroll
      for (int j = 0; j < 4; ++j) {
        unsigned u = __float_as_uint(f[j]);
        float lo = f[j] - __uint_as_float(u & 0xFFFF0000u);
        Bhi[(nq * 4 + j) * LDK + k] = (unsigned short)(u >> 16);
        Blo[(nq * 4 + j) * LDK + k] = f2bf(lo);
      }
    }
    __syncthreads();

    // ---- compute: KC/32 WMMA K-steps ----
    #pragma unroll
    for (int ks = 0; ks < KC; ks += 32) {
      // A fragments (16x32 bf16): lane half0 K=[0..7]+[16..23], half1 K=[8..15]+[24..31]
      bf16x16 ah[2], al[2], bh[4], bl[4];
      #pragma unroll
      for (int ms = 0; ms < 2; ++ms) {
        int rowA = m0 + ms * 16 + l15;
        int kA   = ks + lhalf * 8;
        Frag32B fh, fl;
        fh.q0 = *(const uint4*)&Ahi[rowA * LDK + kA];
        fh.q1 = *(const uint4*)&Ahi[rowA * LDK + kA + 16];
        fl.q0 = *(const uint4*)&Alo[rowA * LDK + kA];
        fl.q1 = *(const uint4*)&Alo[rowA * LDK + kA + 16];
        ah[ms] = __builtin_bit_cast(bf16x16, fh);
        al[ms] = __builtin_bit_cast(bf16x16, fl);
      }
      // B fragments (32x16 bf16): column N=lane%16, half0 K=[0..7]+[16..23], half1 K=[8..15]+[24..31]
      #pragma unroll
      for (int ns = 0; ns < 4; ++ns) {
        int col = n0 + ns * 16 + l15;
        int kB  = ks + lhalf * 8;
        Frag32B gh, gl;
        gh.q0 = *(const uint4*)&Bhi[col * LDK + kB];
        gh.q1 = *(const uint4*)&Bhi[col * LDK + kB + 16];
        gl.q0 = *(const uint4*)&Blo[col * LDK + kB];
        gl.q1 = *(const uint4*)&Blo[col * LDK + kB + 16];
        bh[ns] = __builtin_bit_cast(bf16x16, gh);
        bl[ns] = __builtin_bit_cast(bf16x16, gl);
      }
      // Term-major passes: 8 independent accumulators between reuses -> no
      // TRANS32_DEP stalls on the XDL pipeline.
      #pragma unroll
      for (int ns = 0; ns < 4; ++ns)
        #pragma unroll
        for (int ms = 0; ms < 2; ++ms)
          acc[ms][ns] = __builtin_amdgcn_wmma_f32_16x16x32_bf16(
              false, al[ms], false, bh[ns], (short)0, acc[ms][ns], false, false);
      #pragma unroll
      for (int ns = 0; ns < 4; ++ns)
        #pragma unroll
        for (int ms = 0; ms < 2; ++ms)
          acc[ms][ns] = __builtin_amdgcn_wmma_f32_16x16x32_bf16(
              false, ah[ms], false, bl[ns], (short)0, acc[ms][ns], false, false);
      #pragma unroll
      for (int ns = 0; ns < 4; ++ns)
        #pragma unroll
        for (int ms = 0; ms < 2; ++ms)
          acc[ms][ns] = __builtin_amdgcn_wmma_f32_16x16x32_bf16(
              false, ah[ms], false, bh[ns], (short)0, acc[ms][ns], false, false);
    }
    __syncthreads();
  }

  // ---- epilogue: bias add + store (C layout: M = r + 8*lhalf, N = lane&15) ----
  #pragma unroll
  for (int ns = 0; ns < 4; ++ns) {
    int ncol = n0 + ns * 16 + l15;
    float bv = bg[ncol];
    #pragma unroll
    for (int ms = 0; ms < 2; ++ms) {
      int mbase = m0 + ms * 16 + lhalf * 8;
      #pragma unroll
      for (int r = 0; r < 8; ++r) {
        og[(size_t)(mbase + r) * N_DIM + ncol] = acc[ms][ns][r] + bv;
      }
    }
  }
}

extern "C" void kernel_launch(void* const* d_in, const int* in_sizes, int n_in,
                              void* d_out, int out_size, void* d_ws, size_t ws_size,
                              hipStream_t stream) {
  (void)in_sizes; (void)n_in; (void)out_size; (void)d_ws; (void)ws_size;
  const float* x    = (const float*)d_in[0];
  const int*   sid  = (const int*)d_in[1];
  const float* w    = (const float*)d_in[2];
  const float* bias = (const float*)d_in[3];
  float*       out  = (float*)d_out;

  dim3 grid(B_DIM * (T_DIM / BT) * (N_DIM / BN));   // 256 * 4 * 2 = 2048 blocks
  subject_layers_gemm<<<grid, 256, 0, stream>>>(x, sid, w, bias, out);
}
